// MultiHeadAttention_44263932953359
// MI455X (gfx1250) — compile-verified, matrix operand order checked by reading, not run on hardware
//
#include <hip/hip_runtime.h>
#include <hip/hip_bf16.h>

// ---------------------------------------------------------------------------
// Types for CDNA5 WMMA (wave32): bf16 16x16x32, f32 accumulate.
// ---------------------------------------------------------------------------
typedef __attribute__((ext_vector_type(16))) __bf16 v16bf;
typedef __attribute__((ext_vector_type(8)))  float  v8f;

union FragB {
    v16bf v;
    unsigned short u[16];
    uint4 q[2];
};

__device__ inline v8f wmma_bf16(const FragB& a, const FragB& b, v8f c) {
    // (neg_a, A, neg_b, B, c_mod, C, reuse_a, reuse_b)
    return __builtin_amdgcn_wmma_f32_16x16x32_bf16(false, a.v, false, b.v,
                                                   (short)0, c, false, false);
}

// round-to-nearest-even f32 -> bf16 bits
__device__ inline unsigned short f2bf(float f) {
    unsigned u = __float_as_uint(f);
    u += 0x7FFFu + ((u >> 16) & 1u);
    return (unsigned short)(u >> 16);
}

// ---------------------------------------------------------------------------
// Kernel 0: f32 -> bf16 convert
// ---------------------------------------------------------------------------
__global__ void __launch_bounds__(256) cvt_bf16_kernel(const float* __restrict__ in,
                                                       unsigned short* __restrict__ out,
                                                       int n) {
    int i = blockIdx.x * blockDim.x + threadIdx.x;
    if (i < n) out[i] = f2bf(in[i]);
}

// ---------------------------------------------------------------------------
// Kernel 1: fused QKV projection.
//   C[i,j] = sum_k x[i,k] * W[j,k] + bias[j]
//   One wave computes a 16x64 tile; A fragment reused across 4 N-subtiles.
//   Q, K written bf16 as (b,h,n,d); V written TRANSPOSED as (b,h,d,n) so the
//   attention P.V stage gets contiguous 16B B-fragment loads.
// grid = (128, 48), block = 32
// ---------------------------------------------------------------------------
__global__ void __launch_bounds__(32) qkv_proj_kernel(
    const unsigned short* __restrict__ xb,
    const unsigned short* __restrict__ wqb,
    const unsigned short* __restrict__ wkvb,
    const float* __restrict__ bq,
    const float* __restrict__ bkv,
    unsigned short* __restrict__ qws,
    unsigned short* __restrict__ kws,
    unsigned short* __restrict__ vws) {
    const int i0    = blockIdx.x * 16;   // row tile (M = 2048)
    const int jbase = blockIdx.y * 64;   // col tile (N = 3072)
    const int lane  = threadIdx.x;
    const int half  = lane >> 4;
    const int ln    = lane & 15;

    v8f acc[4] = {};

    for (int k0 = 0; k0 < 1024; k0 += 32) {
        FragB a;
        const unsigned short* arow = xb + (size_t)(i0 + ln) * 1024 + k0 + 8 * half;
        a.q[0] = *(const uint4*)(arow);
        a.q[1] = *(const uint4*)(arow + 16);
#pragma unroll
        for (int t = 0; t < 4; ++t) {
            int j = jbase + t * 16 + ln;
            const unsigned short* wrow =
                (j < 1024) ? (wqb + (size_t)j * 1024)
                           : (wkvb + (size_t)(j - 1024) * 1024);
            FragB bf;
            bf.q[0] = *(const uint4*)(wrow + k0 + 8 * half);
            bf.q[1] = *(const uint4*)(wrow + k0 + 16 + 8 * half);
            acc[t] = wmma_bf16(a, bf, acc[t]);
        }
    }

#pragma unroll
    for (int t = 0; t < 4; ++t) {
        int j = jbase + t * 16 + ln;
        float bias;
        unsigned short* dst;
        int jj;
        if (j < 1024)      { jj = j;        bias = bq[j];         dst = qws; }
        else if (j < 2048) { jj = j - 1024; bias = bkv[j - 1024]; dst = kws; }
        else               { jj = j - 2048; bias = bkv[j - 1024]; dst = vws; }
        const int hh = jj >> 6, dd = jj & 63;
#pragma unroll
        for (int r = 0; r < 8; ++r) {
            int i  = i0 + r + 8 * half;          // C layout: VGPR r -> row r / 8+r
            int bb = i >> 10, nn = i & 1023;
            unsigned short vbf = f2bf(acc[t][r] + bias);
            size_t idx;
            if (j < 2048)
                idx = (((size_t)(bb * 16 + hh)) * 1024 + nn) * 64 + dd;  // (b,h,n,d)
            else
                idx = (((size_t)(bb * 16 + hh)) * 64 + dd) * 1024 + nn;  // (b,h,d,n)
            dst[idx] = vbf;
        }
    }
}

// ---------------------------------------------------------------------------
// Score-tile helper: S(16x16) = Q(16x64) . K(16x64)^T for column tile at mm.
// ---------------------------------------------------------------------------
__device__ inline v8f score_tile(const FragB& a0, const FragB& a1,
                                 const unsigned short* __restrict__ kb,
                                 int mm, int half, int ln) {
    FragB b0, b1;
    const unsigned short* krow = kb + (size_t)(mm + ln) * 64 + 8 * half;
    b0.q[0] = *(const uint4*)(krow);
    b0.q[1] = *(const uint4*)(krow + 16);
    b1.q[0] = *(const uint4*)(krow + 32);
    b1.q[1] = *(const uint4*)(krow + 48);
    v8f s = {};
    s = wmma_bf16(a0, b0, s);
    s = wmma_bf16(a1, b1, s);
    return s;
}

// ---------------------------------------------------------------------------
// Kernel 2: attention. One block = one (b, 16-query-row tile), 16 waves =
// one wave per head.
//   Pass 1: per-wave online softmax stats (registers + shfl reductions).
//   Pass 2: recompute score tiles, stage P in LDS [sub][i][m][h], then
//           (a) fully-coalesced float4 stores of attn (the 134MB output:
//               for fixed row i, the 32m x 16h region is 2KB contiguous),
//           (b) P.V via 4 WMMAs per 32-m chunk (A from LDS, B = contiguous
//               16B loads from transposed V).
// grid = (64, 2), block = 512
// ---------------------------------------------------------------------------
__global__ void __launch_bounds__(512) attn_kernel(
    const unsigned short* __restrict__ qws,
    const unsigned short* __restrict__ kws,
    const unsigned short* __restrict__ vtws,
    float* __restrict__ attnp,
    unsigned short* __restrict__ ows) {
    __shared__ float pls[2][16][16][17];   // [sub][i][m][h], padded

    const int it   = blockIdx.x;   // query tile
    const int b    = blockIdx.y;
    const int tid  = threadIdx.x;
    const int h    = tid >> 5;     // wave id == head
    const int lane = tid & 31;
    const int half = lane >> 4;
    const int ln   = lane & 15;

    const size_t hb = ((size_t)b * 16 + h) * 65536;
    const unsigned short* qb  = qws  + hb;   // (n,d) rows, d contiguous
    const unsigned short* kb  = kws  + hb;   // (n,d) rows, d contiguous
    const unsigned short* vtb = vtws + hb;   // (d,n) rows, n contiguous

    // Q fragments (16 rows x 64 d), resident in VGPRs
    FragB a0, a1;
    {
        const unsigned short* qrow = qb + (size_t)(it * 16 + ln) * 64 + 8 * half;
        a0.q[0] = *(const uint4*)(qrow);
        a0.q[1] = *(const uint4*)(qrow + 16);
        a1.q[0] = *(const uint4*)(qrow + 32);
        a1.q[1] = *(const uint4*)(qrow + 48);
    }

    float rmax[8], rsum[8];
#pragma unroll
    for (int r = 0; r < 8; ++r) { rmax[r] = -__builtin_inff(); rsum[r] = 0.0f; }

    // -------- pass 1: online row max / sum --------
    for (int mm = 0; mm < 1024; mm += 16) {
        v8f s = score_tile(a0, a1, kb, mm, half, ln);
#pragma unroll
        for (int r = 0; r < 8; ++r) {
            float v = s[r] * 0.125f;   // SCALE = 64^-0.5
            float tm = v;
            tm = fmaxf(tm, __shfl_xor(tm, 1));
            tm = fmaxf(tm, __shfl_xor(tm, 2));
            tm = fmaxf(tm, __shfl_xor(tm, 4));
            tm = fmaxf(tm, __shfl_xor(tm, 8));
            float nm = fmaxf(rmax[r], tm);
            float ts = __expf(v - nm);
            ts += __shfl_xor(ts, 1);
            ts += __shfl_xor(ts, 2);
            ts += __shfl_xor(ts, 4);
            ts += __shfl_xor(ts, 8);
            rsum[r] = rsum[r] * __expf(rmax[r] - nm) + ts;
            rmax[r] = nm;
        }
    }
    float rinv[8];
#pragma unroll
    for (int r = 0; r < 8; ++r) rinv[r] = 1.0f / rsum[r];

    // -------- pass 2 --------
    v8f o[4] = {};
    for (int m0 = 0; m0 < 1024; m0 += 32) {
#pragma unroll
        for (int sub = 0; sub < 2; ++sub) {
            v8f s = score_tile(a0, a1, kb, m0 + sub * 16, half, ln);
#pragma unroll
            for (int r = 0; r < 8; ++r) {
                float p = __expf(s[r] * 0.125f - rmax[r]) * rinv[r];
                pls[sub][r + 8 * half][ln][h] = p;
            }
        }
        __syncthreads();

        // (a) coalesced attn output: 16 rows x (32 m x 16 h) = 8192 floats.
        //     Per row i the 512-float (m,h) region is contiguous in memory.
        {
            const size_t base = (((size_t)b * 1024 + it * 16) * 1024 + m0) * 16;
#pragma unroll
            for (int t = 0; t < 4; ++t) {
                int vecIdx = tid + t * 512;      // 0..2047 float4's
                int row    = vecIdx >> 7;        // 128 float4 per row
                int off    = (vecIdx & 127) * 4; // float offset in row region
                int mrel   = off >> 4;           // 0..31
                int h0     = off & 15;           // 0,4,8,12
                const float* src = &pls[mrel >> 4][row][mrel & 15][h0];
                float4 val = make_float4(src[0], src[1], src[2], src[3]);
                *(float4*)(attnp + base + (size_t)row * 16384 + off) = val;
            }
        }

        // (b) P.V accumulation
        FragB pa;
        const int c0 = 8 * half;
#pragma unroll
        for (int z = 0; z < 8; ++z) {
            pa.u[z]     = f2bf(pls[0][ln][c0 + z][h]);  // k chunk [8*half..]
            pa.u[8 + z] = f2bf(pls[1][ln][c0 + z][h]);  // k chunk [16+8*half..]
        }
#pragma unroll
        for (int qd = 0; qd < 4; ++qd) {
            FragB vf;
            const unsigned short* vrow =
                vtb + (size_t)(qd * 16 + ln) * 1024 + m0 + c0;
            vf.q[0] = *(const uint4*)(vrow);        // k = c0..c0+7
            vf.q[1] = *(const uint4*)(vrow + 16);   // k = 16+c0..16+c0+7
            o[qd] = wmma_bf16(pa, vf, o[qd]);
        }
        __syncthreads();
    }

    // store pre-projection output as bf16, (b, n, h*64+d) row-major
#pragma unroll
    for (int qd = 0; qd < 4; ++qd) {
#pragma unroll
        for (int r = 0; r < 8; ++r) {
            int i   = it * 16 + r + 8 * half;
            int col = h * 64 + qd * 16 + ln;
            ows[((size_t)b * 1024 + i) * 1024 + col] = f2bf(o[qd][r]);
        }
    }
}

// ---------------------------------------------------------------------------
// Kernel 3: output projection. C = O_bf16 @ Wp^T + bp, f32 output.
// grid = (128, 16), block = 32
// ---------------------------------------------------------------------------
__global__ void __launch_bounds__(32) out_proj_kernel(
    const unsigned short* __restrict__ ob,
    const unsigned short* __restrict__ wpb,
    const float* __restrict__ bp,
    float* __restrict__ out) {
    const int i0    = blockIdx.x * 16;
    const int jbase = blockIdx.y * 64;
    const int lane  = threadIdx.x;
    const int half  = lane >> 4;
    const int ln    = lane & 15;

    v8f acc[4] = {};

    for (int k0 = 0; k0 < 1024; k0 += 32) {
        FragB a;
        const unsigned short* arow = ob + (size_t)(i0 + ln) * 1024 + k0 + 8 * half;
        a.q[0] = *(const uint4*)(arow);
        a.q[1] = *(const uint4*)(arow + 16);
#pragma unroll
        for (int t = 0; t < 4; ++t) {
            int j = jbase + t * 16 + ln;
            const unsigned short* wrow = wpb + (size_t)j * 1024;
            FragB bf;
            bf.q[0] = *(const uint4*)(wrow + k0 + 8 * half);
            bf.q[1] = *(const uint4*)(wrow + k0 + 16 + 8 * half);
            acc[t] = wmma_bf16(a, bf, acc[t]);
        }
    }

#pragma unroll
    for (int t = 0; t < 4; ++t) {
        int j = jbase + t * 16 + ln;
        float bias = bp[j];
#pragma unroll
        for (int r = 0; r < 8; ++r) {
            int i = i0 + r + 8 * half;
            out[(size_t)i * 1024 + j] = acc[t][r] + bias;
        }
    }
}

// ---------------------------------------------------------------------------
// Host launcher
// ---------------------------------------------------------------------------
extern "C" void kernel_launch(void* const* d_in, const int* in_sizes, int n_in,
                              void* d_out, int out_size, void* d_ws, size_t ws_size,
                              hipStream_t stream) {
    const float* x   = (const float*)d_in[0];   // (2,1024,1024)
    const float* Wq  = (const float*)d_in[1];   // (1024,1024)
    const float* bq  = (const float*)d_in[2];   // (1024,)
    const float* Wkv = (const float*)d_in[3];   // (2048,1024)
    const float* bkv = (const float*)d_in[4];   // (2048,)
    const float* Wp  = (const float*)d_in[5];   // (1024,1024)
    const float* bp  = (const float*)d_in[6];   // (1024,)

    float* outp  = (float*)d_out;                           // 2*1024*1024 f32
    float* attnp = (float*)d_out + (size_t)2 * 1024 * 1024; // 2*1024*1024*16 f32

    // workspace layout (ushort elements)
    unsigned short* xb   = (unsigned short*)d_ws;            // 2M
    unsigned short* wqb  = xb   + (size_t)2 * 1024 * 1024;   // 1M
    unsigned short* wkvb = wqb  + (size_t)1 * 1024 * 1024;   // 2M
    unsigned short* wpb  = wkvb + (size_t)2 * 1024 * 1024;   // 1M
    unsigned short* qws  = wpb  + (size_t)1 * 1024 * 1024;   // 2M  (b,h,n,d)
    unsigned short* kws  = qws  + (size_t)2 * 1024 * 1024;   // 2M  (b,h,n,d)
    unsigned short* vws  = kws  + (size_t)2 * 1024 * 1024;   // 2M  (b,h,d,n)
    unsigned short* ows  = vws  + (size_t)2 * 1024 * 1024;   // 2M  (b,n,c)

    // 1) bf16 conversions
    cvt_bf16_kernel<<<(2097152 + 255) / 256, 256, 0, stream>>>(x,   xb,   2097152);
    cvt_bf16_kernel<<<(1048576 + 255) / 256, 256, 0, stream>>>(Wq,  wqb,  1048576);
    cvt_bf16_kernel<<<(2097152 + 255) / 256, 256, 0, stream>>>(Wkv, wkvb, 2097152);
    cvt_bf16_kernel<<<(1048576 + 255) / 256, 256, 0, stream>>>(Wp,  wpb,  1048576);

    // 2) fused QKV projection
    qkv_proj_kernel<<<dim3(128, 48), 32, 0, stream>>>(xb, wqb, wkvb, bq, bkv,
                                                      qws, kws, vws);

    // 3) attention (softmax probs -> attnp, context -> ows)
    attn_kernel<<<dim3(64, 2), 512, 0, stream>>>(qws, kws, vws, attnp, ows);

    // 4) output projection
    out_proj_kernel<<<dim3(128, 16), 32, 0, stream>>>(ows, wpb, bp, outp);
}
